// DeepICDDescCandiAttention_40922448396637
// MI455X (gfx1250) — matrix operand order repaired and checked by reading, not compiled, not run on hardware
//
#include <hip/hip_runtime.h>
#include <hip/hip_bf16.h>

// ---------------------------------------------------------------------------
// Problem constants (from the reference)
// ---------------------------------------------------------------------------
#define B_  16
#define L_  2048
#define D_  1024
#define OS_ 64
#define NC_ 256
#define LAB_ 1024
#define BN_EPS_ 1e-5f

typedef __attribute__((ext_vector_type(16))) __bf16 v16bf;
typedef __attribute__((ext_vector_type(8)))  __bf16 v8bf;
typedef __attribute__((ext_vector_type(4)))  __bf16 v4bf;
typedef __attribute__((ext_vector_type(8)))  float  v8f;

#define LDT 40   // LDS tile row stride in halfs (32 data + 8 pad -> conflict-free b128)

__device__ __forceinline__ v4bf cvt4(float4 f) {
    v4bf h; h[0] = (__bf16)f.x; h[1] = (__bf16)f.y; h[2] = (__bf16)f.z; h[3] = (__bf16)f.w;
    return h;
}

// ---------------------------------------------------------------------------
// Batched GEMM, bf16 WMMA with f32 accumulate, 2-stage software pipeline with
// double-buffered LDS tiles.
//   C[b, m, n] = epilogue( sum_k A(b,m,k) * B(b,k,n) )
//   A(b,m,k)           = A[b*sAb + m*sAm + k]              (k-contiguous fp32)
//   B(k,n)  non-gather = Bm[b*sBb + k*sBk + n]             (n-contiguous fp32)
//   B(k,n)  gather     = Bm[bidx[b*N+n]*sBrow + k]         (k-contiguous fp32)
//   Epilogue: +bias[n], BN(gamma,beta,mean,var over n), ReLU.
// Block tile 64x64, BK=32, 128 threads = 4 waves. Wave w: rows [16w,16w+16),
// all 64 cols -> 4 WMMA accumulators.
// LDS: At[m][k] row-major; Bt[n][k] TRANSPOSED so each lane's fragment is a
// contiguous run -> ds_load_b128 fragment fetches, rows padded to 40 halfs.
// ---------------------------------------------------------------------------
__global__ __launch_bounds__(128)
void gemm_bf16_wmma(const float* __restrict__ A, long sAb, long sAm,
                    const float* __restrict__ Bm, long sBb, long sBk, long sBrow,
                    const int* __restrict__ bidx,
                    float* __restrict__ C, long sCb, int ldc,
                    int N, int K,
                    const float* __restrict__ bias,
                    const float* __restrict__ bn_g, const float* __restrict__ bn_b,
                    const float* __restrict__ bn_m, const float* __restrict__ bn_v,
                    int do_relu)
{
    const int b   = blockIdx.z;
    const int m0g = blockIdx.y << 6;
    const int n0g = blockIdx.x << 6;

    __shared__ __bf16 At[2][64 * LDT];   // [m][k]
    __shared__ __bf16 Bt[2][64 * LDT];   // [n][k]  (transposed B tile)

    const int tid  = threadIdx.x;
    const int wave = tid >> 5;
    const int lane = tid & 31;
    const int lg   = lane >> 4;          // lane group (0: lanes 0-15, 1: 16-31)
    const int r16  = lane & 15;

    v8f acc[4];
    #pragma unroll
    for (int t = 0; t < 4; ++t)
        #pragma unroll
        for (int r = 0; r < 8; ++r) acc[t][r] = 0.0f;

    const long aBase = (long)b * sAb;
    const long bBase = (long)b * sBb;

    // ---- per-thread staging coordinates (constant across K loop) ----
    const int am  = tid >> 3;            // A row, +16*i
    const int ak  = (tid * 4) & 31;      // A k within tile
    const int bgn = tid >> 3;            // gather-B row (n), +16*i
    const int bgk = (tid * 4) & 31;      // gather-B k
    const int bnn = (tid * 4) & 63;      // non-gather-B n (4 consecutive)
    const int bnk = tid >> 4;            // non-gather-B k, +8*i

    long growb[4] = {0, 0, 0, 0};        // hoisted gather row bases
    if (bidx) {
        #pragma unroll
        for (int i = 0; i < 4; ++i)
            growb[i] = (long)bidx[(long)b * N + n0g + bgn + 16 * i] * sBrow;
    }

    float4 ra[4], rb[4];

    auto loadTiles = [&](int kb) {       // issue global loads, no wait
        #pragma unroll
        for (int i = 0; i < 4; ++i)
            ra[i] = *(const float4*)&A[aBase + (long)(m0g + am + 16 * i) * sAm + (kb + ak)];
        if (bidx) {
            #pragma unroll
            for (int i = 0; i < 4; ++i)
                rb[i] = *(const float4*)&Bm[growb[i] + (kb + bgk)];
        } else {
            #pragma unroll
            for (int i = 0; i < 4; ++i)
                rb[i] = *(const float4*)&Bm[bBase + (long)(kb + bnk + 8 * i) * sBk + (n0g + bnn)];
        }
    };

    auto storeTiles = [&](int buf) {     // cvt fp32->bf16 and fill LDS buffer
        #pragma unroll
        for (int i = 0; i < 4; ++i)
            *(v4bf*)&At[buf][(am + 16 * i) * LDT + ak] = cvt4(ra[i]);
        if (bidx) {
            #pragma unroll
            for (int i = 0; i < 4; ++i)
                *(v4bf*)&Bt[buf][(bgn + 16 * i) * LDT + bgk] = cvt4(rb[i]);
        } else {
            #pragma unroll
            for (int i = 0; i < 4; ++i) {
                const int k = bnk + 8 * i;
                Bt[buf][(bnn + 0) * LDT + k] = (__bf16)rb[i].x;
                Bt[buf][(bnn + 1) * LDT + k] = (__bf16)rb[i].y;
                Bt[buf][(bnn + 2) * LDT + k] = (__bf16)rb[i].z;
                Bt[buf][(bnn + 3) * LDT + k] = (__bf16)rb[i].w;
            }
        }
    };

    // ---- prologue: stage tile 0 ----
    loadTiles(0);
    storeTiles(0);

    int cur = 0;
    for (int kb = 0; kb < K; kb += 32) {
        __syncthreads();
        const bool more = (kb + 32) < K;
        if (more) loadTiles(kb + 32);                 // next tile: issue loads early
        if (kb + 64 < K)                              // warm L2 two tiles ahead
            __builtin_prefetch(&A[aBase + (long)(m0g + am) * sAm + (kb + 64 + ak)], 0, 1);

        // ---- compute from buf[cur] ----
        // A fragment (16x32 bf16, ISA 7.12.2): lane (r16,lg) needs
        // K = lg*8 + 0..7 and 16 + lg*8 + 0..7 -> two contiguous 16B runs.
        const __bf16* at = &At[cur][((wave << 4) + r16) * LDT + (lg << 3)];
        const v8bf alo = *(const v8bf*)(at);
        const v8bf ahi = *(const v8bf*)(at + 16);

        // B fragments: lane needs col n = t*16+r16, K = lg*8 + 0..7 and
        // 16 + lg*8 + 0..7 -> two contiguous 16B runs. Load all four before any WMMA.
        v8bf bl[4], bh[4];
        #pragma unroll
        for (int t = 0; t < 4; ++t) {
            const __bf16* bt = &Bt[cur][((t << 4) + r16) * LDT + (lg << 3)];
            bl[t] = *(const v8bf*)(bt);
            bh[t] = *(const v8bf*)(bt + 16);
        }

        const v16bf af = __builtin_shufflevector(alo, ahi,
                             0, 1, 2, 3, 4, 5, 6, 7, 8, 9, 10, 11, 12, 13, 14, 15);
        #pragma unroll
        for (int t = 0; t < 4; ++t) {
            const v16bf bf = __builtin_shufflevector(bl[t], bh[t],
                                 0, 1, 2, 3, 4, 5, 6, 7, 8, 9, 10, 11, 12, 13, 14, 15);
            acc[t] = __builtin_amdgcn_wmma_f32_16x16x32_bf16(
                         false, af, false, bf, (short)0, acc[t], false, false);
        }

        if (more) storeTiles(cur ^ 1);                // wait loads, cvt, fill other buf
        cur ^= 1;
    }

    // ---- epilogue + store (C layout: VGPR r -> M = r + 8*lg, N = r16) ----
    const int mw = m0g + (wave << 4) + (lg << 3);
    #pragma unroll
    for (int t = 0; t < 4; ++t) {
        const int n = n0g + (t << 4) + r16;
        const float bia = bias ? bias[n] : 0.0f;
        float sc = 1.0f, sh = 0.0f;
        if (bn_g) {
            const float iv = rsqrtf(bn_v[n] + BN_EPS_);
            sc = bn_g[n] * iv;
            sh = bn_b[n] - bn_m[n] * sc;
        }
        #pragma unroll
        for (int r = 0; r < 8; ++r) {
            float x = (acc[t][r] + bia) * sc + sh;
            if (do_relu) x = fmaxf(x, 0.0f);
            C[(long)b * sCb + (long)(mw + r) * ldc + n] = x;
        }
    }
}

// ---------------------------------------------------------------------------
// Softmax over L (2048) for each (b, o); writes TRANSPOSED alpha [B][OS][L].
// ---------------------------------------------------------------------------
__global__ __launch_bounds__(256)
void softmax_over_L(const float* __restrict__ S, float* __restrict__ AT)
{
    const int bo = blockIdx.x;
    const int b = bo >> 6, o = bo & 63;
    const float* col = S + (long)b * L_ * OS_ + o;        // stride OS_
    float* dst = AT + ((long)b * OS_ + o) * L_;

    __shared__ float red[256];
    float v[8];
    float mx = -3.4e38f;
    #pragma unroll
    for (int i = 0; i < 8; ++i) {
        const int l = threadIdx.x + (i << 8);
        v[i] = col[(long)l * OS_];
        mx = fmaxf(mx, v[i]);
    }
    red[threadIdx.x] = mx; __syncthreads();
    for (int s = 128; s > 0; s >>= 1) {
        if (threadIdx.x < s) red[threadIdx.x] = fmaxf(red[threadIdx.x], red[threadIdx.x + s]);
        __syncthreads();
    }
    mx = red[0]; __syncthreads();

    float sum = 0.0f;
    #pragma unroll
    for (int i = 0; i < 8; ++i) { v[i] = __expf(v[i] - mx); sum += v[i]; }
    red[threadIdx.x] = sum; __syncthreads();
    for (int s = 128; s > 0; s >>= 1) {
        if (threadIdx.x < s) red[threadIdx.x] += red[threadIdx.x + s];
        __syncthreads();
    }
    const float inv = 1.0f / red[0];
    #pragma unroll
    for (int i = 0; i < 8; ++i) dst[threadIdx.x + (i << 8)] = v[i] * inv;
}

// ---------------------------------------------------------------------------
// Softmax over OS (64) for each (b, c); writes TRANSPOSED alpha2 [B][NC][OS].
// ---------------------------------------------------------------------------
__global__ __launch_bounds__(64)
void softmax_over_O(const float* __restrict__ S2, float* __restrict__ A2T)
{
    const int bc = blockIdx.x;
    const int b = bc >> 8, c = bc & 255;
    const int o = threadIdx.x;

    __shared__ float red[64];
    const float x = S2[((long)b * OS_ + o) * NC_ + c];
    red[o] = x; __syncthreads();
    for (int s = 32; s > 0; s >>= 1) { if (o < s) red[o] = fmaxf(red[o], red[o + s]); __syncthreads(); }
    const float mx = red[0]; __syncthreads();
    const float e = __expf(x - mx);
    red[o] = e; __syncthreads();
    for (int s = 32; s > 0; s >>= 1) { if (o < s) red[o] += red[o + s]; __syncthreads(); }
    A2T[((long)b * NC_ + c) * OS_ + o] = e / red[0];
}

// ---------------------------------------------------------------------------
// Launch pipeline
// ---------------------------------------------------------------------------
extern "C" void kernel_launch(void* const* d_in, const int* in_sizes, int n_in,
                              void* d_out, int out_size, void* d_ws, size_t ws_size,
                              hipStream_t stream)
{
    (void)in_sizes; (void)n_in; (void)out_size; (void)ws_size;

    const float* X    = (const float*)d_in[0];
    const int*   cand = (const int*)d_in[1];     // jax default x64-off -> int32
    const float* a_w  = (const float*)d_in[2];
    const float* a_b  = (const float*)d_in[3];
    const float* h_w  = (const float*)d_in[4];
    const float* h_b  = (const float*)d_in[5];
    const float* bn_g = (const float*)d_in[6];
    const float* bn_b = (const float*)d_in[7];
    const float* bn_m = (const float*)d_in[8];
    const float* bn_v = (const float*)d_in[9];
    const float* lab  = (const float*)d_in[10];
    float* out = (float*)d_out;

    float* ws  = (float*)d_ws;
    float* s1  = ws;                              // [B][L][OS]   scores1
    float* aT  = s1  + (size_t)B_ * L_ * OS_;     // [B][OS][L]   alpha^T
    float* Xp  = aT  + (size_t)B_ * OS_ * L_;     // [B][OS][D]
    float* Xp2 = Xp  + (size_t)B_ * OS_ * D_;     // [B][OS][LAB] (post BN+ReLU)
    float* s2  = Xp2 + (size_t)B_ * OS_ * LAB_;   // [B][OS][NC]  scores2
    float* a2T = s2  + (size_t)B_ * OS_ * NC_;    // [B][NC][OS]  alpha2^T

    const dim3 blk(128);

    // 1) scores1 = X @ a_w + a_b            [B, L, OS]
    gemm_bf16_wmma<<<dim3(OS_ / 64, L_ / 64, B_), blk, 0, stream>>>(
        X, (long)L_ * D_, D_,
        a_w, 0, OS_, 0, nullptr,
        s1, (long)L_ * OS_, OS_, OS_, D_,
        a_b, nullptr, nullptr, nullptr, nullptr, 0);

    // 2) alpha^T = softmax_L(scores1)       [B, OS, L]
    softmax_over_L<<<B_ * OS_, 256, 0, stream>>>(s1, aT);

    // 3) Xp = alpha^T @ X                   [B, OS, D]
    gemm_bf16_wmma<<<dim3(D_ / 64, OS_ / 64, B_), blk, 0, stream>>>(
        aT, (long)OS_ * L_, L_,
        X, (long)L_ * D_, D_, 0, nullptr,
        Xp, (long)OS_ * D_, D_, D_, L_,
        nullptr, nullptr, nullptr, nullptr, nullptr, 0);

    // 4) Xp2 = relu(BN(Xp @ h_w + h_b))     [B, OS, LAB]
    gemm_bf16_wmma<<<dim3(LAB_ / 64, OS_ / 64, B_), blk, 0, stream>>>(
        Xp, (long)OS_ * D_, D_,
        h_w, 0, LAB_, 0, nullptr,
        Xp2, (long)OS_ * LAB_, LAB_, LAB_, D_,
        h_b, bn_g, bn_b, bn_m, bn_v, 1);

    // 5) scores2 = Xp2 @ labDescVec[cand]^T [B, OS, NC]   (row gather on B)
    gemm_bf16_wmma<<<dim3(NC_ / 64, OS_ / 64, B_), blk, 0, stream>>>(
        Xp2, (long)OS_ * LAB_, LAB_,
        lab, 0, 0, LAB_, cand,
        s2, (long)OS_ * NC_, NC_, NC_, LAB_,
        nullptr, nullptr, nullptr, nullptr, nullptr, 0);

    // 6) alpha2^T = softmax_OS(scores2)     [B, NC, OS]
    softmax_over_O<<<B_ * NC_, 64, 0, stream>>>(s2, a2T);

    // 7) out = alpha2^T @ Xp2               [B, NC, LAB]
    gemm_bf16_wmma<<<dim3(LAB_ / 64, NC_ / 64, B_), blk, 0, stream>>>(
        a2T, (long)NC_ * OS_, OS_,
        Xp2, (long)OS_ * LAB_, LAB_, 0, nullptr,
        out, (long)NC_ * LAB_, LAB_, LAB_, OS_,
        nullptr, nullptr, nullptr, nullptr, nullptr, 0);
}
